// GraphCare_40295383171396
// MI455X (gfx1250) — compile-verified, matrix operand order checked by reading, not run on hardware
//
#include <hip/hip_runtime.h>
#include <hip/hip_bf16.h>

typedef __attribute__((ext_vector_type(16))) _Float16 v16h;
typedef __attribute__((ext_vector_type(8)))  _Float16 h8v;
typedef __attribute__((ext_vector_type(8)))  float    v8f;

#define B_   32
#define T_   20
#define V_   1024
#define EMB_ 128
#define EPS_ 1e-6f

// ------------------------------------------------------------------
// Generic batched WMMA GEMM:  C[z] = act( A16[z][M,K] @ B16[z][N,K]^T + bias[z] )
// One wave owns a 32x64 tile: 2 A frags, 4 B frags, 8 v_wmma per K-step of 32.
// ------------------------------------------------------------------
struct GemmArgs {
  const _Float16* A[4];
  const _Float16* Bm[4];
  float*          C[4];
  const float*    bias[4];
  int M, N, K, act;   // act: 0 none, 1 relu, 2 tanh
};

__device__ __forceinline__ v16h load_frag(const _Float16* q) {
  // lane<16: rows M=lane, K = kh..kh+7 and kh+16..kh+23 (kh folded into q by caller)
  union { v16h v; h8v p[2]; } u;
  u.p[0] = *(const h8v*)q;
  u.p[1] = *(const h8v*)(q + 16);
  return u.v;
}

__global__ void k_gemm(GemmArgs g) {
  const int z = blockIdx.z;
  const _Float16* A  = g.A[z];
  const _Float16* Bm = g.Bm[z];
  float*          C  = g.C[z];
  const float* bias  = g.bias[z];
  const int K = g.K, N = g.N;
  const int lane = threadIdx.x & 31;
  const int wave = threadIdx.x >> 5;
  const int wpb  = blockDim.x >> 5;
  const int rowBase = blockIdx.y << 5;                       // 32 rows per wave tile
  const int colBase = (blockIdx.x * wpb + wave) << 6;        // 64 cols per wave tile
  const int frow = lane & 15;
  const int kh   = (lane >> 4) << 3;        // 0 or 8 per WMMA 16-bit layout
  const _Float16* Ap0 = A  + (size_t)(rowBase +  0 + frow) * K + kh;
  const _Float16* Ap1 = A  + (size_t)(rowBase + 16 + frow) * K + kh;
  const _Float16* Bp0 = Bm + (size_t)(colBase +  0 + frow) * K + kh;
  const _Float16* Bp1 = Bm + (size_t)(colBase + 16 + frow) * K + kh;
  const _Float16* Bp2 = Bm + (size_t)(colBase + 32 + frow) * K + kh;
  const _Float16* Bp3 = Bm + (size_t)(colBase + 48 + frow) * K + kh;
  v8f acc00 = {}, acc01 = {}, acc02 = {}, acc03 = {};
  v8f acc10 = {}, acc11 = {}, acc12 = {}, acc13 = {};
  for (int k = 0; k < K; k += 32) {
    v16h a0 = load_frag(Ap0 + k);
    v16h a1 = load_frag(Ap1 + k);
    v16h b0 = load_frag(Bp0 + k);
    v16h b1 = load_frag(Bp1 + k);
    v16h b2 = load_frag(Bp2 + k);
    v16h b3 = load_frag(Bp3 + k);
    acc00 = __builtin_amdgcn_wmma_f32_16x16x32_f16(false, a0, false, b0, (short)0, acc00, false, false);
    acc01 = __builtin_amdgcn_wmma_f32_16x16x32_f16(false, a0, false, b1, (short)0, acc01, false, false);
    acc02 = __builtin_amdgcn_wmma_f32_16x16x32_f16(false, a0, false, b2, (short)0, acc02, false, false);
    acc03 = __builtin_amdgcn_wmma_f32_16x16x32_f16(false, a0, false, b3, (short)0, acc03, false, false);
    acc10 = __builtin_amdgcn_wmma_f32_16x16x32_f16(false, a1, false, b0, (short)0, acc10, false, false);
    acc11 = __builtin_amdgcn_wmma_f32_16x16x32_f16(false, a1, false, b1, (short)0, acc11, false, false);
    acc12 = __builtin_amdgcn_wmma_f32_16x16x32_f16(false, a1, false, b2, (short)0, acc12, false, false);
    acc13 = __builtin_amdgcn_wmma_f32_16x16x32_f16(false, a1, false, b3, (short)0, acc13, false, false);
  }
  // C/D layout: VGPR m -> row base + (lane>>4)*8 + m ; col = colBase + (lane&15) + 16*nt
  const int cCol = colBase + (lane & 15);
  float bv0 = bias[cCol], bv1 = bias[cCol + 16], bv2 = bias[cCol + 32], bv3 = bias[cCol + 48];
  #pragma unroll
  for (int r = 0; r < 2; r++) {
    const int cRow = rowBase + (r << 4) + ((lane >> 4) << 3);
    v8f a0 = r ? acc10 : acc00;
    v8f a1 = r ? acc11 : acc01;
    v8f a2 = r ? acc12 : acc02;
    v8f a3 = r ? acc13 : acc03;
    #pragma unroll
    for (int m = 0; m < 8; m++) {
      float o0 = a0[m] + bv0, o1 = a1[m] + bv1, o2 = a2[m] + bv2, o3 = a3[m] + bv3;
      if (g.act == 1) { o0 = fmaxf(o0, 0.f); o1 = fmaxf(o1, 0.f); o2 = fmaxf(o2, 0.f); o3 = fmaxf(o3, 0.f); }
      else if (g.act == 2) { o0 = tanhf(o0); o1 = tanhf(o1); o2 = tanhf(o2); o3 = tanhf(o3); }
      float* crow = C + (size_t)(cRow + m) * N + cCol;
      crow[0] = o0; crow[16] = o1; crow[32] = o2; crow[48] = o3;
    }
  }
}

// ---------------- elementwise / prep kernels ----------------
__global__ void k_cvt16(const float* s, _Float16* d, int n) {
  for (int i = blockIdx.x * blockDim.x + threadIdx.x; i < n; i += gridDim.x * blockDim.x)
    d[i] = (_Float16)s[i];
}

__global__ void k_fill_u32(unsigned* p, unsigned v, size_t n) {
  for (size_t i = blockIdx.x * (size_t)blockDim.x + threadIdx.x; i < n;
       i += (size_t)gridDim.x * blockDim.x) p[i] = v;
}

__global__ void k_build_rx(const float* visc, const float* visp, _Float16* rx16) {
  int idx = blockIdx.x * blockDim.x + threadIdx.x;          // 2*640*1024
  if (idx >= 2 * 640 * 1024) return;
  int branch = idx >= 640 * 1024;
  int r = idx - branch * 640 * 1024;
  int v = r & 1023;
  int row = r >> 10;                                        // b*20 + t
  int b = row / T_, t = row % T_;
  const float* src = branch ? visp : visc;
  rx16[idx] = (_Float16)src[(b * T_ + (T_ - 1 - t)) * V_ + v];
}

__global__ void k_gru_gate(const float* xg, const float* gh, float* h, _Float16* h16,
                           float* outs, _Float16* outs16, int t) {
  int idx = blockIdx.x * blockDim.x + threadIdx.x;          // 4*32*1024
  if (idx >= 4 * 32 * 1024) return;
  int inst = idx >> 15;
  int b    = (idx >> 10) & 31;
  int j    = idx & 1023;
  const float* xgp = xg + ((size_t)inst * 640 + b * T_ + t) * 3072;
  const float* ghp = gh + ((size_t)inst * 32 + b) * 3072;
  float xr = xgp[j], xz = xgp[1024 + j], xn = xgp[2048 + j];
  float hr = ghp[j], hz = ghp[1024 + j], hn = ghp[2048 + j];
  float rg = 1.f / (1.f + expf(-(xr + hr)));
  float zg = 1.f / (1.f + expf(-(xz + hz)));
  float ng = tanhf(xn + rg * hn);
  size_t hoff = ((size_t)inst * 32 + b) * 1024 + j;
  float hnew = (1.f - zg) * ng + zg * h[hoff];
  h[hoff] = hnew;  h16[hoff] = (_Float16)hnew;
  size_t ooff = ((size_t)inst * 640 + b * T_ + t) * 1024 + j;
  outs[ooff] = hnew;  outs16[ooff] = (_Float16)hnew;
}

__global__ void k_alpha_logit(const float* outs, const float* wC, const float* bC,
                              const float* wP, const float* bP, float* alogit) {
  int row  = blockIdx.x * (blockDim.x >> 5) + (threadIdx.x >> 5);  // 0..1279
  int lane = threadIdx.x & 31;
  if (row >= 1280) return;
  int branch = row >= 640;
  int r = row - branch * 640;
  const float* g = outs + ((size_t)(branch * 2) * 640 + r) * 1024;  // inst 0 or 2
  const float* w = branch ? wP : wC;
  float s = 0.f;
  for (int j = lane; j < 1024; j += 32) s += g[j] * w[j];
  for (int off = 16; off; off >>= 1) s += __shfl_xor(s, off, 32);
  if (lane == 0) alogit[row] = s + (branch ? bP[0] : bC[0]);
}

__global__ void k_softmax(const float* alogit, float* alpha) {
  int bb = blockIdx.x;                      // branch*32 + b
  int lane = threadIdx.x;
  int base = (bb >> 5) * 640 + (bb & 31) * T_;
  float v = (lane < T_) ? alogit[base + lane] : -INFINITY;
  float m = v;
  for (int off = 16; off; off >>= 1) m = fmaxf(m, __shfl_xor(m, off, 32));
  float e = (lane < T_) ? expf(v - m) : 0.f;
  float s = e;
  for (int off = 16; off; off >>= 1) s += __shfl_xor(s, off, 32);
  if (lane < T_) alpha[base + lane] = e / s;
}

__global__ void k_attn(const float* alpha, const float* beta,
                       const float* visc, const float* visp, float* attn) {
  int idx = blockIdx.x * blockDim.x + threadIdx.x;          // 32*1024
  if (idx >= 32 * 1024) return;
  int b = idx >> 10, v = idx & 1023;
  float s = 0.f;
  for (int t = 0; t < T_; t++) {
    int row = b * T_ + t;
    float rc = visc[(b * T_ + (T_ - 1 - t)) * V_ + v];
    float rp = visp[(b * T_ + (T_ - 1 - t)) * V_ + v];
    s += alpha[row]       * beta[(size_t)row * 1024 + v] * rc;
    s += alpha[640 + row] * beta[(size_t)(640 + row) * 1024 + v] * rp;
  }
  attn[idx] = s;
}

__device__ __forceinline__ void atomicMaxF(float* addr, float val) {
  unsigned* ua = (unsigned*)addr;
  unsigned old = *ua;
  while (__uint_as_float(old) < val) {
    unsigned assumed = old;
    old = atomicCAS(ua, assumed, __float_as_uint(val));
    if (old == assumed) break;
  }
}

__global__ void k_attnw_max(const int* batch, const int* ids, const float* attn,
                            float* attn_w, float* maxcell, int N) {
  int i = blockIdx.x * blockDim.x + threadIdx.x;
  float w = -INFINITY;
  if (i < N) { w = attn[batch[i] * V_ + ids[i]]; attn_w[i] = w; }
  for (int off = 16; off; off >>= 1) w = fmaxf(w, __shfl_xor(w, off, 32));
  if ((threadIdx.x & 31) == 0) atomicMaxF(maxcell, w);
}

__global__ void k_attnw_norm(float* attn_w, const float* maxcell, int N) {
  int i = blockIdx.x * blockDim.x + threadIdx.x;
  if (i >= N) return;
  float w = attn_w[i] / maxcell[0];
  if (w != w) w = 0.f;
  attn_w[i] = w;
}

__global__ void k_edge(const int* ei, const float* attn_w, float* ea, int E) {
  int e = blockIdx.x * blockDim.x + threadIdx.x;
  if (e >= E) return;
  ea[e] = attn_w[ei[e]] + attn_w[ei[E + e]] + 2.f * EPS_;
}

__global__ void k_gather_x0(const int* ids, const float* emb, float* x, int N) {
  int idx = blockIdx.x * blockDim.x + threadIdx.x;          // N*32
  if (idx >= N * 32) return;
  int i = idx >> 5, c4 = (idx & 31) << 2;
  *(float4*)(x + (size_t)i * EMB_ + c4) = *(const float4*)(emb + (size_t)ids[i] * EMB_ + c4);
}

__global__ void k_scatter(const int* ei, const float* ea, const float* x,
                          const float* eW, const float* eb, float* agg, int E) {
  size_t idx = blockIdx.x * (size_t)blockDim.x + threadIdx.x;   // E*32
  if (idx >= (size_t)E * 32) return;
  int e = (int)(idx >> 5), c4 = ((int)idx & 31) << 2;
  int s = ei[e], d = ei[E + e];
  float a = ea[e];
  float4 xv = *(const float4*)(x + (size_t)s * EMB_ + c4);
  float4 wv = *(const float4*)(eW + c4);
  float4 bv = *(const float4*)(eb + c4);
  float m0 = fmaxf(xv.x + a * wv.x + bv.x, 0.f);
  float m1 = fmaxf(xv.y + a * wv.y + bv.y, 0.f);
  float m2 = fmaxf(xv.z + a * wv.z + bv.z, 0.f);
  float m3 = fmaxf(xv.w + a * wv.w + bv.w, 0.f);
  float* ag = agg + (size_t)d * EMB_ + c4;
  atomicAdd(ag + 0, m0); atomicAdd(ag + 1, m1);
  atomicAdd(ag + 2, m2); atomicAdd(ag + 3, m3);
}

__global__ void k_xsum16(const float* x, const float* agg, _Float16* out, size_t n) {
  for (size_t i = blockIdx.x * (size_t)blockDim.x + threadIdx.x; i < n;
       i += (size_t)gridDim.x * blockDim.x) out[i] = (_Float16)(x[i] + agg[i]);
}

__global__ void k_pidx(const int* ids, const int* patient, int* pidx, int N) {
  int i = blockIdx.x * blockDim.x + threadIdx.x;
  if (i >= N) return;
  int v = ids[i];
  for (int b = 0; b < B_; b++)
    if (v == patient[b]) atomicMin(&pidx[b], i);
}

__global__ void k_xp(const float* x, const int* pidx, _Float16* xp16) {
  int idx = blockIdx.x * blockDim.x + threadIdx.x;          // 32*128
  if (idx >= B_ * EMB_) return;
  int b = idx >> 7, c = idx & 127;
  xp16[idx] = (_Float16)x[(size_t)pidx[b] * EMB_ + c];
}

// ------------------------------------------------------------------
extern "C" void kernel_launch(void* const* d_in, const int* in_sizes, int n_in,
                              void* d_out, int out_size, void* d_ws, size_t ws_size,
                              hipStream_t stream) {
  if (n_in < 45) return;
  // ---- input mapping (jax tree_flatten: sorted dict keys) ----
  const int*   node_ids = (const int*)  d_in[0];
  const int*   edge_idx = (const int*)  d_in[1];
  const int*   batch    = (const int*)  d_in[2];
  const float* visc     = (const float*)d_in[3];
  const float* visp     = (const float*)d_in[4];
  const int*   patient  = (const int*)  d_in[5];
  const float* emb      = (const float*)d_in[6];
  const int CB = 7, PB = 33;   // cond / proc bases
  // retain params offsets: +0 aWhh, +1 aWih, +2 abhh, +3 abih, +4 aliW, +5 alib,
  //                        +6 bWhh, +7 bWih, +8 bbhh, +9 bbih, +10 bliW, +11 blib
  const float* convW[3] = {(const float*)d_in[19], (const float*)d_in[23], (const float*)d_in[27]};
  const float* convB[3] = {(const float*)d_in[20], (const float*)d_in[24], (const float*)d_in[28]};
  const float* conveW[3]= {(const float*)d_in[21], (const float*)d_in[25], (const float*)d_in[29]};
  const float* conveB[3]= {(const float*)d_in[22], (const float*)d_in[26], (const float*)d_in[30]};
  const float* fcW = (const float*)d_in[31];
  const float* fcB = (const float*)d_in[32];
  const int N = in_sizes[0];
  const int E = in_sizes[1] / 2;

  // ---- workspace bump allocator ----
  size_t off = 0;
  auto bump = [&](size_t bytes) -> char* {
    off = (off + 255) & ~(size_t)255;
    char* p = (char*)d_ws + off;
    off += bytes;
    return p;
  };
  // persistent
  float*    attn    = (float*)bump(32 * 1024 * 4);
  float*    attn_w  = (float*)bump((size_t)N * 4);
  float*    maxcell = (float*)bump(256);
  float*    ea      = (float*)bump((size_t)E * 4);
  _Float16* convW16 = (_Float16*)bump(4 * 128 * 128 * 2);
  int*      pidx    = (int*)bump(256);
  _Float16* xp16    = (_Float16*)bump(B_ * EMB_ * 2);
  size_t unionBase = off;
  // retain scratch
  _Float16* rx16   = (_Float16*)bump((size_t)2 * 640 * 1024 * 2);
  _Float16* Wih16  = (_Float16*)bump((size_t)4 * 3072 * 1024 * 2);
  _Float16* Whh16  = (_Float16*)bump((size_t)4 * 3072 * 1024 * 2);
  _Float16* bliW16 = (_Float16*)bump((size_t)2 * 1024 * 1024 * 2);
  float*    xg     = (float*)bump((size_t)4 * 640 * 3072 * 4);
  float*    gh     = (float*)bump((size_t)4 * 32 * 3072 * 4);
  float*    hst    = (float*)bump((size_t)4 * 32 * 1024 * 4);
  _Float16* h16    = (_Float16*)bump((size_t)4 * 32 * 1024 * 2);
  float*    outs   = (float*)bump((size_t)4 * 640 * 1024 * 4);
  _Float16* outs16 = (_Float16*)bump((size_t)4 * 640 * 1024 * 2);
  float*    alogit = (float*)bump(1280 * 4);
  float*    alpha  = (float*)bump(1280 * 4);
  float*    beta   = (float*)bump((size_t)2 * 640 * 1024 * 4);
  // gine scratch (aliases retain scratch; phases are stream-ordered)
  off = unionBase;
  float*    xA     = (float*)bump((size_t)N * EMB_ * 4);
  float*    xB     = (float*)bump((size_t)N * EMB_ * 4);
  float*    agg    = (float*)bump((size_t)N * EMB_ * 4);
  _Float16* xsum16 = (_Float16*)bump((size_t)N * EMB_ * 2);

  auto cvt = [&](const float* s, _Float16* d, int n) {
    int blocks = (n + 255) / 256; if (blocks > 4096) blocks = 4096;
    k_cvt16<<<blocks, 256, 0, stream>>>(s, d, n);
  };
  auto fill = [&](void* p, unsigned v, size_t n) {
    size_t blocks = (n + 255) / 256; if (blocks > 4096) blocks = 4096;
    k_fill_u32<<<(unsigned)blocks, 256, 0, stream>>>((unsigned*)p, v, n);
  };
  auto launch_gemm = [&](GemmArgs& ga, int nbatch) {
    int wpb = (ga.N % 256 == 0) ? 4 : 2;
    dim3 blk(wpb * 32, 1, 1);
    dim3 grd(ga.N / (64 * wpb), ga.M / 32, nbatch);
    k_gemm<<<grd, blk, 0, stream>>>(ga);
  };

  // ---- prep: weight conversions ----
  const int GW = 3072 * 1024;
  cvt((const float*)d_in[CB + 1], Wih16 + 0 * GW, GW);  // cond.aWih
  cvt((const float*)d_in[CB + 7], Wih16 + 1 * GW, GW);  // cond.bWih
  cvt((const float*)d_in[PB + 1], Wih16 + 2 * GW, GW);  // proc.aWih
  cvt((const float*)d_in[PB + 7], Wih16 + 3 * GW, GW);  // proc.bWih
  cvt((const float*)d_in[CB + 0], Whh16 + 0 * GW, GW);
  cvt((const float*)d_in[CB + 6], Whh16 + 1 * GW, GW);
  cvt((const float*)d_in[PB + 0], Whh16 + 2 * GW, GW);
  cvt((const float*)d_in[PB + 6], Whh16 + 3 * GW, GW);
  cvt((const float*)d_in[CB + 10], bliW16,               1024 * 1024);
  cvt((const float*)d_in[PB + 10], bliW16 + 1024 * 1024, 1024 * 1024);
  for (int l = 0; l < 3; l++) cvt(convW[l], convW16 + l * 16384, 16384);
  cvt(fcW, convW16 + 3 * 16384, 16384);
  k_build_rx<<<(2 * 640 * 1024 + 255) / 256, 256, 0, stream>>>(visc, visp, rx16);
  fill(hst, 0u, 4 * 32 * 1024);
  fill(h16, 0u, 4 * 32 * 1024 / 2);

  // ---- xg = rx @ Wih^T + bih  (batched over 4 GRUs) ----
  GemmArgs gx{};
  gx.A[0] = rx16;             gx.A[1] = rx16;
  gx.A[2] = rx16 + 640*1024;  gx.A[3] = rx16 + 640*1024;
  for (int i = 0; i < 4; i++) { gx.Bm[i] = Wih16 + (size_t)i * GW; gx.C[i] = xg + (size_t)i * 640 * 3072; }
  gx.bias[0] = (const float*)d_in[CB + 3]; gx.bias[1] = (const float*)d_in[CB + 9];
  gx.bias[2] = (const float*)d_in[PB + 3]; gx.bias[3] = (const float*)d_in[PB + 9];
  gx.M = 640; gx.N = 3072; gx.K = 1024; gx.act = 0;
  launch_gemm(gx, 4);

  // ---- GRU recurrence: 20 sequential steps, 4 GRUs batched ----
  GemmArgs gr{};
  for (int i = 0; i < 4; i++) {
    gr.A[i]  = h16 + (size_t)i * 32 * 1024;
    gr.Bm[i] = Whh16 + (size_t)i * GW;
    gr.C[i]  = gh + (size_t)i * 32 * 3072;
  }
  gr.bias[0] = (const float*)d_in[CB + 2]; gr.bias[1] = (const float*)d_in[CB + 8];
  gr.bias[2] = (const float*)d_in[PB + 2]; gr.bias[3] = (const float*)d_in[PB + 8];
  gr.M = 32; gr.N = 3072; gr.K = 1024; gr.act = 0;
  for (int t = 0; t < T_; t++) {
    launch_gemm(gr, 4);
    k_gru_gate<<<(4 * 32 * 1024) / 256, 256, 0, stream>>>(xg, gh, hst, h16, outs, outs16, t);
  }

  // ---- alpha ----
  k_alpha_logit<<<320, 128, 0, stream>>>(outs,
      (const float*)d_in[CB + 4], (const float*)d_in[CB + 5],
      (const float*)d_in[PB + 4], (const float*)d_in[PB + 5], alogit);
  k_softmax<<<64, 32, 0, stream>>>(alogit, alpha);

  // ---- beta = tanh(h @ bliW^T + blib)  (batched over 2 branches) ----
  GemmArgs gb{};
  gb.A[0]  = outs16 + (size_t)1 * 640 * 1024;  gb.A[1]  = outs16 + (size_t)3 * 640 * 1024;
  gb.Bm[0] = bliW16;                           gb.Bm[1] = bliW16 + 1024 * 1024;
  gb.C[0]  = beta;                             gb.C[1]  = beta + (size_t)640 * 1024;
  gb.bias[0] = (const float*)d_in[CB + 11];    gb.bias[1] = (const float*)d_in[PB + 11];
  gb.M = 640; gb.N = 1024; gb.K = 1024; gb.act = 2;
  launch_gemm(gb, 2);

  // ---- attn / attn_w / edges ----
  k_attn<<<(32 * 1024) / 256, 256, 0, stream>>>(alpha, beta, visc, visp, attn);
  fill(maxcell, 0xFF800000u, 1);
  k_attnw_max<<<(N + 255) / 256, 256, 0, stream>>>(batch, node_ids, attn, attn_w, maxcell, N);
  k_attnw_norm<<<(N + 255) / 256, 256, 0, stream>>>(attn_w, maxcell, N);
  k_edge<<<(E + 255) / 256, 256, 0, stream>>>(edge_idx, attn_w, ea, E);

  // ---- GINE x3 ----
  k_gather_x0<<<(N * 32 + 255) / 256, 256, 0, stream>>>(node_ids, emb, xA, N);
  float* xin = xA; float* xout = xB;
  for (int l = 0; l < 3; l++) {
    fill(agg, 0u, (size_t)N * EMB_);
    k_scatter<<<(unsigned)(((size_t)E * 32 + 255) / 256), 256, 0, stream>>>(
        edge_idx, ea, xin, conveW[l], conveB[l], agg, E);
    k_xsum16<<<4096, 256, 0, stream>>>(xin, agg, xsum16, (size_t)N * EMB_);
    GemmArgs gg{};
    gg.A[0] = xsum16; gg.Bm[0] = convW16 + l * 16384; gg.C[0] = xout; gg.bias[0] = convB[l];
    gg.M = N; gg.N = 128; gg.K = 128; gg.act = 1;
    launch_gemm(gg, 1);
    float* tmp = xin; xin = xout; xout = tmp;
  }

  // ---- final readout ----
  fill(pidx, 0x7FFFFFFFu, 64);
  k_pidx<<<(N + 255) / 256, 256, 0, stream>>>(node_ids, patient, pidx, N);
  k_xp<<<(B_ * EMB_ + 255) / 256, 256, 0, stream>>>(xin, pidx, xp16);
  GemmArgs gf{};
  gf.A[0] = xp16; gf.Bm[0] = convW16 + 3 * 16384; gf.C[0] = (float*)d_out; gf.bias[0] = fcB;
  gf.M = 32; gf.N = 128; gf.K = 128; gf.act = 0;
  launch_gemm(gf, 1);
}
